// ReMMTAS_85727547228460
// MI455X (gfx1250) — matrix operand order repaired and checked by reading, not compiled, not run on hardware
//
#include <hip/hip_runtime.h>

// ---------------- problem constants (from reference) ----------------
#define DI   256   // D_IN (also layer d_in and DV for both layers)
#define HH   8     // heads
#define BB   2     // batch
#define SS   2048  // sequence
#define SEGL 512
#define STL  128
#define LF   768   // ST + SEG + ST
#define DK   128
#define DV   256
#define NSEG 4

typedef _Float16 v16h __attribute__((ext_vector_type(16)));
typedef _Float16 v8h  __attribute__((ext_vector_type(8)));
typedef float    v8f  __attribute__((ext_vector_type(8)));

// ---------------- WMMA fragment loaders (CDNA5 16x16x32 f16 layouts) ----------------
// A (16x32, MxK): lane m=lane&15; lanes0-15: K = [0..7],[16..23]; lanes16-31: K = [8..15],[24..31]
__device__ __forceinline__ v16h frag_a(const _Float16* __restrict__ base, long lda, int lane, int kc) {
  const _Float16* p = base + (long)(lane & 15) * lda + kc + ((lane >> 4) << 3);
  v8h lo = *(const v8h*)(p);
  v8h hi = *(const v8h*)(p + 16);
  return __builtin_shufflevector(lo, hi, 0,1,2,3,4,5,6,7,8,9,10,11,12,13,14,15);
}
// B (32x16, KxN) fed from Bt (NxK row-major): lane n=lane&15; lanes0-15: K = [0..7],[16..23]; lanes16-31: K = [8..15],[24..31]
__device__ __forceinline__ v16h frag_b(const _Float16* __restrict__ base, long ldb, int lane, int kc) {
  const _Float16* p = base + (long)(lane & 15) * ldb + kc + ((lane >> 4) << 3);
  v8h lo = *(const v8h*)(p);
  v8h hi = *(const v8h*)(p + 16);
  return __builtin_shufflevector(lo, hi, 0,1,2,3,4,5,6,7,8,9,10,11,12,13,14,15);
}

#define WMMA_F16(a, b, c) \
  __builtin_amdgcn_wmma_f32_16x16x32_f16(false, (a), false, (b), (short)0, (c), false, false)

// ---------------- generic batched GEMM: C[b,g] = alpha * sum_ic A[b,g,ic] * Bt[b,g,ic]^T ----------
struct GemmP {
  const _Float16* A;   // M x K rows (lda), pre-offset to section
  const _Float16* Bt;  // N x K rows (ldb)  (i.e. B transposed)
  void* C;
  int M, N, K;
  long lda, ldb, ldc;
  long sAb, sAg, sBb, sBg, sCb, sCg;   // batch (b) and group (g) strides
  int G;                               // groups per batch (z = b*G + g)
  int inner;                           // inner summation count (heads for _sl)
  long iSA, iSB;                       // inner strides
  float alpha;
};

// OMODE: 0 = f32 row-major, 1 = f16 row-major, 2 = f16 transposed (C[n*ldc + m])
template <int OMODE>
__global__ __launch_bounds__(128) void k_gemm(GemmP p) {
  const int lane = threadIdx.x & 31;
  const int wave = threadIdx.x >> 5;
  const int z = blockIdx.z;
  const int b = z / p.G;
  const int g = z - b * p.G;
  const int mt = blockIdx.x * 32;
  const int nb = blockIdx.y * 128 + wave * 32;
  if (nb >= p.N) return;  // wave-uniform

  const _Float16* A0 = p.A + (long)b * p.sAb + (long)g * p.sAg + (long)mt * p.lda;
  const _Float16* B0 = p.Bt + (long)b * p.sBb + (long)g * p.sBg + (long)nb * p.ldb;

  v8f acc00 = {0.f,0.f,0.f,0.f,0.f,0.f,0.f,0.f};
  v8f acc01 = acc00, acc10 = acc00, acc11 = acc00;

  for (int ic = 0; ic < p.inner; ++ic) {
    const _Float16* Ai = A0 + (long)ic * p.iSA;
    const _Float16* Bi = B0 + (long)ic * p.iSB;
    for (int kc = 0; kc < p.K; kc += 32) {
      v16h a0 = frag_a(Ai, p.lda, lane, kc);
      v16h a1 = frag_a(Ai + 16 * p.lda, p.lda, lane, kc);
      v16h b0 = frag_b(Bi, p.ldb, lane, kc);
      v16h b1 = frag_b(Bi + 16 * p.ldb, p.ldb, lane, kc);
      acc00 = WMMA_F16(a0, b0, acc00);
      acc01 = WMMA_F16(a0, b1, acc01);
      acc10 = WMMA_F16(a1, b0, acc10);
      acc11 = WMMA_F16(a1, b1, acc11);
    }
  }

  const int rbase = (lane >> 4) << 3;   // C: VGPR r -> row r + 8*(lane>>4)
  const int col = lane & 15;
  if (OMODE == 0) {
    float* C = (float*)p.C + (long)b * p.sCb + (long)g * p.sCg;
    #pragma unroll
    for (int mi = 0; mi < 2; ++mi) {
      v8f a0 = mi ? acc10 : acc00;
      v8f a1 = mi ? acc11 : acc01;
      long r0 = mt + 16 * mi + rbase;
      #pragma unroll
      for (int r = 0; r < 8; ++r) {
        C[(r0 + r) * p.ldc + nb + col]      = a0[r] * p.alpha;
        C[(r0 + r) * p.ldc + nb + 16 + col] = a1[r] * p.alpha;
      }
    }
  } else if (OMODE == 1) {
    _Float16* C = (_Float16*)p.C + (long)b * p.sCb + (long)g * p.sCg;
    #pragma unroll
    for (int mi = 0; mi < 2; ++mi) {
      v8f a0 = mi ? acc10 : acc00;
      v8f a1 = mi ? acc11 : acc01;
      long r0 = mt + 16 * mi + rbase;
      #pragma unroll
      for (int r = 0; r < 8; ++r) {
        C[(r0 + r) * p.ldc + nb + col]      = (_Float16)(a0[r] * p.alpha);
        C[(r0 + r) * p.ldc + nb + 16 + col] = (_Float16)(a1[r] * p.alpha);
      }
    }
  } else {
    _Float16* C = (_Float16*)p.C + (long)b * p.sCb + (long)g * p.sCg;
    #pragma unroll
    for (int mi = 0; mi < 2; ++mi) {
      #pragma unroll
      for (int ni = 0; ni < 2; ++ni) {
        v8f ac = mi ? (ni ? acc11 : acc10) : (ni ? acc01 : acc00);
        long r0 = mt + 16 * mi + rbase;
        int cc = nb + 16 * ni + col;
        v8h hv;
        #pragma unroll
        for (int r = 0; r < 8; ++r) hv[r] = (_Float16)(ac[r] * p.alpha);
        *(v8h*)(C + (long)cc * p.ldc + r0) = hv;   // 8 contiguous rows -> 16B store
      }
    }
  }
}

// ---------------- elementwise / helper kernels ----------------
// W[h][g][d][e] (f32) -> wT[h][g][e][d] (f16)
__global__ __launch_bounds__(256) void k_convert_w(const float* __restrict__ W,
                                                   _Float16* __restrict__ wT,
                                                   int HG, int Din, int E) {
  long idx = (long)blockIdx.x * blockDim.x + threadIdx.x;
  long per = (long)Din * E;
  if (idx >= (long)HG * per) return;
  long hg = idx / per;
  long rem = idx - hg * per;
  int d = (int)(rem / E);
  int e = (int)(rem - (long)d * E);
  wT[hg * per + (long)e * Din + d] = (_Float16)W[idx];
}

__global__ __launch_bounds__(256) void k_init_state(const float* __restrict__ init,
                                                    _Float16* __restrict__ st) {
  int idx = blockIdx.x * blockDim.x + threadIdx.x;
  const int per = STL * DI;
  if (idx >= BB * per) return;
  st[idx] = (_Float16)init[idx % per];
}

__global__ __launch_bounds__(256) void k_build_h0(const float* __restrict__ x,
                                                  const _Float16* __restrict__ st,
                                                  _Float16* __restrict__ h0, int seg) {
  long idx = (long)blockIdx.x * blockDim.x + threadIdx.x;
  if (idx >= (long)BB * HH * LF * DI) return;
  int d = (int)(idx % DI);
  long t = idx / DI;
  int l = (int)(t % LF);
  t /= LF;                 // t = b*HH + g
  int b = (int)(t / HH);
  _Float16 v;
  if (l < STL)             v = st[((long)b * STL + l) * DI + d];
  else if (l < STL + SEGL) v = (_Float16)x[((long)b * SS + seg * SEGL + (l - STL)) * DI + d];
  else                     v = st[((long)b * STL + (l - STL - SEGL)) * DI + d];
  h0[idx] = v;
}

__global__ __launch_bounds__(256) void k_rope(_Float16* __restrict__ q, int offset) {
  long idx = (long)blockIdx.x * blockDim.x + threadIdx.x;
  if (idx >= (long)BB * HH * LF * (DK / 2)) return;
  int ip = (int)(idx % (DK / 2));
  long bl = idx / (DK / 2);
  int l = (int)(bl % LF);
  _Float16* p = q + bl * DK + 2 * ip;
  float pos = (float)(offset + l);
  float freq = __expf(-(float)(2 * ip) * (9.210340371976184f / (float)DK)); // 10000^(-2i/D)
  float a = pos * freq;
  float c = __cosf(a), sn = __sinf(a);
  float t1 = (float)p[0], t2 = (float)p[1];
  p[0] = (_Float16)(t1 * c - t2 * sn);
  p[1] = (_Float16)(t1 * sn + t2 * c);
}

__global__ __launch_bounds__(256) void k_softmax(const float* __restrict__ S,
                                                 _Float16* __restrict__ P) {
  int row = blockIdx.x;                 // (b*H + h)*LF + i
  int i = row % LF;
  const float* s = S + (long)row * LF;
  _Float16* p = P + (long)row * LF;
  int t = threadIdx.x;                  // 0..255, LF = 3*256
  float v[3], e[3];
  float mx = -1e30f;
  #pragma unroll
  for (int r = 0; r < 3; ++r) {
    int j = t + r * 256;
    v[r] = (j <= i) ? s[j] : -1e30f;
    mx = fmaxf(mx, v[r]);
  }
  __shared__ float red[256];
  red[t] = mx; __syncthreads();
  for (int off = 128; off > 0; off >>= 1) {
    if (t < off) red[t] = fmaxf(red[t], red[t + off]);
    __syncthreads();
  }
  mx = red[0]; __syncthreads();
  float sum = 0.f;
  #pragma unroll
  for (int r = 0; r < 3; ++r) {
    int j = t + r * 256;
    e[r] = (j <= i) ? __expf(v[r] - mx) : 0.f;
    sum += e[r];
  }
  red[t] = sum; __syncthreads();
  for (int off = 128; off > 0; off >>= 1) {
    if (t < off) red[t] += red[t + off];
    __syncthreads();
  }
  float rs = 1.f / red[0];
  #pragma unroll
  for (int r = 0; r < 3; ++r) p[t + r * 256] = (_Float16)(e[r] * rs);
}

__global__ __launch_bounds__(256) void k_mean(const float* __restrict__ hout,
                                              float* __restrict__ hm,
                                              _Float16* __restrict__ hmh) {
  long idx = (long)blockIdx.x * blockDim.x + threadIdx.x;
  if (idx >= (long)BB * LF * DI) return;
  int d = (int)(idx % DI);
  long t = idx / DI;
  int l = (int)(t % LF);
  int b = (int)(t / LF);
  float s = 0.f;
  #pragma unroll
  for (int g = 0; g < HH; ++g) s += hout[(((long)b * HH + g) * LF + l) * DI + d];
  s *= 0.125f;
  hm[idx] = s;
  hmh[idx] = (_Float16)s;
}

__global__ __launch_bounds__(256) void k_state_prep(const float* __restrict__ hm,
                                                    _Float16* __restrict__ sp) {
  long idx = (long)blockIdx.x * blockDim.x + threadIdx.x;
  if (idx >= (long)BB * STL * DI) return;
  int d = (int)(idx % DI);
  long t = idx / DI;
  int l = (int)(t % STL);
  int b = (int)(t / STL);
  float a = hm[((long)b * LF + l) * DI + d];
  float c = hm[((long)b * LF + (STL + SEGL) + l) * DI + d];
  sp[idx] = (_Float16)(0.5f * (a + c));
}

// ---------------- host side ----------------
static inline void launch_gemm(int omode, const GemmP& p, hipStream_t st) {
  dim3 grid((unsigned)(p.M / 32), (unsigned)((p.N + 127) / 128), (unsigned)(BB * p.G));
  dim3 blk(128);
  if (omode == 0)      k_gemm<0><<<grid, blk, 0, st>>>(p);
  else if (omode == 1) k_gemm<1><<<grid, blk, 0, st>>>(p);
  else                 k_gemm<2><<<grid, blk, 0, st>>>(p);
}

struct Ctx {
  _Float16 *wq[2][3], *wk[2][3], *wv[2][3], *winv[3];
  _Float16 *woutT, *wstateT;
  _Float16 *state, *hA, *hB, *Qh, *Kh, *Vt, *Pm, *hmh, *sp;
  float *Sc, *houtF, *hmF;
  hipStream_t st;
};

static const int g_l0[3] = {0, STL, STL + SEGL};
static const int g_ls[3] = {STL, SEGL, STL};

// one attention pass for layer `li` on f16 input hin (B,H,LF,DI); output via omode into outC
static void attn_pass(const Ctx& c, int li, const _Float16* hin, int offset,
                      int omode, void* outC) {
  for (int sec = 0; sec < 3; ++sec) {
    int l0 = g_l0[sec], ls = g_ls[sec];
    GemmP p{};
    p.M = ls; p.K = DI; p.G = HH; p.inner = HH; p.alpha = 1.f;
    p.lda = DI; p.sAb = (long)HH * LF * DI; p.sAg = 0; p.iSA = (long)LF * DI;
    p.A = hin + (long)l0 * DI;
    // Q
    p.N = DK; p.ldb = DI; p.sBb = 0; p.sBg = (long)DK * DI; p.iSB = (long)HH * DK * DI;
    p.Bt = c.wq[li][sec];
    p.C = c.Qh + (long)l0 * DK; p.ldc = DK; p.sCb = (long)HH * LF * DK; p.sCg = (long)LF * DK;
    launch_gemm(1, p, c.st);
    // K
    p.Bt = c.wk[li][sec];
    p.C = c.Kh + (long)l0 * DK;
    launch_gemm(1, p, c.st);
    // V -> transposed f16 (B,H,DV,LF)
    p.N = DV; p.sBg = (long)DV * DI; p.iSB = (long)HH * DV * DI;
    p.Bt = c.wv[li][sec];
    p.C = c.Vt + l0; p.ldc = LF; p.sCb = (long)HH * DV * LF; p.sCg = (long)DV * LF;
    launch_gemm(2, p, c.st);
  }
  long nro = (long)BB * HH * LF * (DK / 2);
  k_rope<<<dim3((unsigned)((nro + 255) / 256)), 256, 0, c.st>>>(c.Qh, offset);
  k_rope<<<dim3((unsigned)((nro + 255) / 256)), 256, 0, c.st>>>(c.Kh, offset);
  {
    GemmP p{};
    p.A = c.Qh; p.lda = DK; p.sAb = (long)HH * LF * DK; p.sAg = (long)LF * DK;
    p.Bt = c.Kh; p.ldb = DK; p.sBb = (long)HH * LF * DK; p.sBg = (long)LF * DK;
    p.C = c.Sc; p.ldc = LF; p.sCb = (long)HH * LF * LF; p.sCg = (long)LF * LF;
    p.M = LF; p.N = LF; p.K = DK; p.G = HH; p.inner = 1; p.iSA = 0; p.iSB = 0;
    p.alpha = 0.08838834764831845f;   // 1/sqrt(128)
    launch_gemm(0, p, c.st);
  }
  k_softmax<<<dim3((unsigned)(BB * HH * LF)), 256, 0, c.st>>>(c.Sc, c.Pm);
  {
    GemmP p{};
    p.A = c.Pm; p.lda = LF; p.sAb = (long)HH * LF * LF; p.sAg = (long)LF * LF;
    p.Bt = c.Vt; p.ldb = LF; p.sBb = (long)HH * DV * LF; p.sBg = (long)DV * LF;
    p.C = outC; p.ldc = DV; p.sCb = (long)HH * LF * DV; p.sCg = (long)LF * DV;
    p.M = LF; p.N = DV; p.K = LF; p.G = HH; p.inner = 1; p.iSA = 0; p.iSB = 0; p.alpha = 1.f;
    launch_gemm(omode, p, c.st);
  }
}

extern "C" void kernel_launch(void* const* d_in, const int* in_sizes, int n_in,
                              void* d_out, int out_size, void* d_ws, size_t ws_size,
                              hipStream_t stream) {
  (void)in_sizes; (void)n_in; (void)out_size; (void)ws_size;
  const float* x          = (const float*)d_in[0];
  const float* init_state = (const float*)d_in[22];
  const float* W_out      = (const float*)d_in[23];
  const float* W_state    = (const float*)d_in[24];

  Ctx c{};
  c.st = stream;
  char* cur = (char*)d_ws;
  auto alloc = [&](size_t bytes) -> void* {
    void* r = (void*)cur;
    cur += (bytes + 255) & ~(size_t)255;
    return r;
  };
  const size_t QKW = 64ul * DI * DK;   // halves per q/k weight
  const size_t VW  = 64ul * DI * DV;   // halves per v / winv weight
  for (int li = 0; li < 2; ++li)
    for (int sec = 0; sec < 3; ++sec) {
      c.wq[li][sec] = (_Float16*)alloc(QKW * 2);
      c.wk[li][sec] = (_Float16*)alloc(QKW * 2);
      c.wv[li][sec] = (_Float16*)alloc(VW * 2);
    }
  for (int sec = 0; sec < 3; ++sec) c.winv[sec] = (_Float16*)alloc(VW * 2);
  c.woutT   = (_Float16*)alloc((size_t)DV * DI * 2);
  c.wstateT = (_Float16*)alloc((size_t)DV * DI * 2);
  c.state = (_Float16*)alloc((size_t)BB * STL * DI * 2);
  c.hA    = (_Float16*)alloc((size_t)BB * HH * LF * DI * 2);
  c.hB    = (_Float16*)alloc((size_t)BB * HH * LF * DI * 2);
  c.Qh    = (_Float16*)alloc((size_t)BB * HH * LF * DK * 2);
  c.Kh    = (_Float16*)alloc((size_t)BB * HH * LF * DK * 2);
  c.Vt    = (_Float16*)alloc((size_t)BB * HH * DV * LF * 2);
  c.Pm    = (_Float16*)alloc((size_t)BB * HH * LF * LF * 2);
  c.hmh   = (_Float16*)alloc((size_t)BB * LF * DI * 2);
  c.sp    = (_Float16*)alloc((size_t)BB * STL * DI * 2);
  c.Sc    = (float*)alloc((size_t)BB * HH * LF * LF * 4);
  c.houtF = (float*)alloc((size_t)BB * HH * LF * DV * 4);
  c.hmF   = (float*)alloc((size_t)BB * LF * DI * 4);

  auto conv = [&](const float* src, _Float16* dst, int HG, int Din, int E) {
    long total = (long)HG * Din * E;
    k_convert_w<<<dim3((unsigned)((total + 255) / 256)), 256, 0, stream>>>(src, dst, HG, Din, E);
  };
  // d_in index maps (dict insertion order): layer0: Wq=1,Wk=2,Wv=3,Wq_s=4,Wk_s=5,Wv_s=6,
  // Wq_e=7,Wk_e=8,Wv_e=9, Winv=10,Winv_b=11,Winv_e=12; layer1 base 13 (no Winv).
  const int qi0[3] = {4, 1, 7},   ki0[3] = {5, 2, 8},   vi0[3] = {6, 3, 9};
  const int qi1[3] = {16, 13, 19}, ki1[3] = {17, 14, 20}, vi1[3] = {18, 15, 21};
  const int wiv[3] = {11, 10, 12};  // section order: _b, mid, _e
  for (int sec = 0; sec < 3; ++sec) {
    conv((const float*)d_in[qi0[sec]], c.wq[0][sec], 64, DI, DK);
    conv((const float*)d_in[ki0[sec]], c.wk[0][sec], 64, DI, DK);
    conv((const float*)d_in[vi0[sec]], c.wv[0][sec], 64, DI, DV);
    conv((const float*)d_in[qi1[sec]], c.wq[1][sec], 64, DI, DK);
    conv((const float*)d_in[ki1[sec]], c.wk[1][sec], 64, DI, DK);
    conv((const float*)d_in[vi1[sec]], c.wv[1][sec], 64, DI, DV);
    conv((const float*)d_in[wiv[sec]], c.winv[sec], 64, DV, DI);
  }
  conv(W_out,   c.woutT,   1, DV, DI);
  conv(W_state, c.wstateT, 1, DV, DI);

  k_init_state<<<dim3((BB * STL * DI + 255) / 256), 256, 0, stream>>>(init_state, c.state);

  float* out = (float*)d_out;
  const long h0n = (long)BB * HH * LF * DI;

  for (int s = 0; s < NSEG; ++s) {
    int offset = s * SEGL;
    k_build_h0<<<dim3((unsigned)((h0n + 255) / 256)), 256, 0, stream>>>(x, c.state, c.hA, s);

    // layer0, iter 1 of 2
    attn_pass(c, 0, c.hA, offset, 1, c.hB);
    // inverse projections: hB -> hA  (sections: Winv_b / Winv / Winv_e)
    for (int sec = 0; sec < 3; ++sec) {
      int l0 = g_l0[sec], ls = g_ls[sec];
      GemmP p{};
      p.M = ls; p.N = DI; p.K = DV; p.G = HH; p.inner = HH; p.alpha = 1.f;
      p.A = c.hB + (long)l0 * DV; p.lda = DV; p.sAb = (long)HH * LF * DV; p.sAg = 0;
      p.iSA = (long)LF * DV;
      p.Bt = c.winv[sec]; p.ldb = DV; p.sBb = 0; p.sBg = (long)DI * DV;
      p.iSB = (long)HH * DI * DV;
      p.C = c.hA + (long)l0 * DI; p.ldc = DI; p.sCb = (long)HH * LF * DI; p.sCg = (long)LF * DI;
      launch_gemm(1, p, stream);
    }
    // layer0, iter 2 of 2
    attn_pass(c, 0, c.hA, offset, 1, c.hB);
    // layer1 (f32 output for head-mean)
    attn_pass(c, 1, c.hB, offset, 0, c.houtF);

    k_mean<<<dim3((unsigned)(((long)BB * LF * DI + 255) / 256)), 256, 0, stream>>>(
        c.houtF, c.hmF, c.hmh);
    k_state_prep<<<dim3((BB * STL * DI + 255) / 256), 256, 0, stream>>>(c.hmF, c.sp);

    // out_seg = hm[:, ST:-ST] @ W_out  -> d_out[b, s*SEG : (s+1)*SEG, :]
    {
      GemmP p{};
      p.A = c.hmh + (long)STL * DI; p.lda = DI; p.sAb = (long)LF * DI; p.sAg = 0;
      p.Bt = c.woutT; p.ldb = DI; p.sBb = 0; p.sBg = 0;
      p.C = out + (long)s * SEGL * DI; p.ldc = DI; p.sCb = (long)SS * DI; p.sCg = 0;
      p.M = SEGL; p.N = DI; p.K = DV; p.G = 1; p.inner = 1; p.iSA = 0; p.iSB = 0; p.alpha = 1.f;
      launch_gemm(0, p, stream);
    }
    // new_state = 0.5*(hm[:,:ST]+hm[:,-ST:]) @ W_state  (f16, consumed by next segment)
    {
      GemmP p{};
      p.A = c.sp; p.lda = DI; p.sAb = (long)STL * DI; p.sAg = 0;
      p.Bt = c.wstateT; p.ldb = DI; p.sBb = 0; p.sBg = 0;
      p.C = c.state; p.ldc = DI; p.sCb = (long)STL * DI; p.sCg = 0;
      p.M = STL; p.N = DI; p.K = DV; p.G = 1; p.inner = 1; p.iSA = 0; p.iSB = 0; p.alpha = 1.f;
      launch_gemm(1, p, stream);
    }
  }
}
